// get_model_48988396978316
// MI455X (gfx1250) — matrix operand order repaired and checked by reading, not compiled, not run on hardware
//
#include <hip/hip_runtime.h>
#include <math.h>

// ---------------------------------------------------------------------------
// Types for WMMA
// ---------------------------------------------------------------------------
typedef __attribute__((ext_vector_type(16))) _Float16 v16h;
typedef __attribute__((ext_vector_type(8)))  float    v8f;
typedef __attribute__((ext_vector_type(4)))  float    v4f;

#define BB 8
#define NN 4096

// ---------------------------------------------------------------------------
// Fused GEMM (+bias, optional BN+ReLU) using V_WMMA_F32_16X16X32_F16.
// X: [M, Cin] row-major f32, W: [Cout, Cin] row-major f32.
// Y[m][n] = act( (sum_k X[m][k]*W[n][k]) + bias[n] )  with optional BN.
// Each wave computes TWO adjacent 16x16 N-tiles sharing one A fragment
// (halves A traffic); 4 waves per block -> 64 rows x 32 cols per block.
// OOB rows/cols are clamped for loads and masked at the store, so the K-main
// loop is branch-free. ALIGNED variant (Cin%4==0, 16B-aligned buffers) uses
// float4 (b128) loads.
// ---------------------------------------------------------------------------
template<bool ALIGNED>
__global__ void gemm_bn_relu_k(const float* __restrict__ X,
                               const float* __restrict__ W,
                               const float* __restrict__ bias,
                               const float* __restrict__ bng,
                               const float* __restrict__ bnb,
                               const float* __restrict__ bnm,
                               const float* __restrict__ bnv,
                               float* __restrict__ Y,
                               int M, int Cin, int Cout, int fuse)
{
    const int lane = threadIdx.x & 31;
    const int wave = threadIdx.x >> 5;
    const int mbase  = (blockIdx.x * 4 + wave) * 16;
    const int nbase0 = blockIdx.y * 32;
    const int nbase1 = nbase0 + 16;
    const int hl = lane >> 4;      // wave half
    const int rr = lane & 15;

    int arow  = mbase  + rr; if (arow  >= M)    arow  = M - 1;
    int bcol0 = nbase0 + rr; if (bcol0 >= Cout) bcol0 = Cout - 1;
    int bcol1 = nbase1 + rr; if (bcol1 >= Cout) bcol1 = Cout - 1;

    const float* xrow  = X + (size_t)arow  * Cin;
    const float* wrow0 = W + (size_t)bcol0 * Cin;
    const float* wrow1 = W + (size_t)bcol1 * Cin;

    v8f acc0 = (v8f){0.f, 0.f, 0.f, 0.f, 0.f, 0.f, 0.f, 0.f};
    v8f acc1 = acc0;

    const int kmain = Cin & ~31;
    for (int k0 = 0; k0 < kmain; k0 += 32) {
        v16h a, b0, b1;
        if (ALIGNED) {
            const v4f* xa  = (const v4f*)(xrow  + k0 + hl * 16);
            const v4f* wb0 = (const v4f*)(wrow0 + k0 + hl * 16);
            const v4f* wb1 = (const v4f*)(wrow1 + k0 + hl * 16);
            v4f A0 = xa[0], A1 = xa[1], A2 = xa[2], A3 = xa[3];
            v4f P0 = wb0[0], P1 = wb0[1], P2 = wb0[2], P3 = wb0[3];
            v4f Q0 = wb1[0], Q1 = wb1[1], Q2 = wb1[2], Q3 = wb1[3];
#pragma unroll
            for (int j = 0; j < 4; ++j) {
                a[j]      = (_Float16)A0[j];
                a[4 + j]  = (_Float16)A1[j];
                a[8 + j]  = (_Float16)A2[j];
                a[12 + j] = (_Float16)A3[j];
                b0[j]      = (_Float16)P0[j];
                b0[4 + j]  = (_Float16)P1[j];
                b0[8 + j]  = (_Float16)P2[j];
                b0[12 + j] = (_Float16)P3[j];
                b1[j]      = (_Float16)Q0[j];
                b1[4 + j]  = (_Float16)Q1[j];
                b1[8 + j]  = (_Float16)Q2[j];
                b1[12 + j] = (_Float16)Q3[j];
            }
        } else {
            const float* xa  = xrow  + k0 + hl * 16;    // A: 16 consecutive floats
            const float* wb0 = wrow0 + k0 + hl * 16;    // B: 16 consecutive floats
            const float* wb1 = wrow1 + k0 + hl * 16;
#pragma unroll
            for (int j = 0; j < 8; ++j) {
                a[j]     = (_Float16)xa[j];
                a[8 + j] = (_Float16)xa[8 + j];
            }
#pragma unroll
            for (int j = 0; j < 16; ++j) {
                b0[j] = (_Float16)wb0[j];
                b1[j] = (_Float16)wb1[j];
            }
        }
        acc0 = __builtin_amdgcn_wmma_f32_16x16x32_f16(false, a, false, b0,
                                                      (short)0, acc0, false, false);
        acc1 = __builtin_amdgcn_wmma_f32_16x16x32_f16(false, a, false, b1,
                                                      (short)0, acc1, false, false);
    }
    if (kmain < Cin) {  // ragged K tail: clamped loads + branch-free selects
        v16h a, b0, b1;
#pragma unroll
        for (int j = 0; j < 8; ++j) {
            int k1 = kmain + hl * 16 + j;
            int k2 = k1 + 8;
            float v1 = xrow[k1 < Cin ? k1 : Cin - 1];
            float v2 = xrow[k2 < Cin ? k2 : Cin - 1];
            a[j]     = (_Float16)(k1 < Cin ? v1 : 0.f);
            a[8 + j] = (_Float16)(k2 < Cin ? v2 : 0.f);
        }
#pragma unroll
        for (int j = 0; j < 16; ++j) {
            int k = kmain + hl * 16 + j;
            int kc = k < Cin ? k : Cin - 1;
            float p = wrow0[kc];
            float q = wrow1[kc];
            b0[j] = (_Float16)(k < Cin ? p : 0.f);
            b1[j] = (_Float16)(k < Cin ? q : 0.f);
        }
        acc0 = __builtin_amdgcn_wmma_f32_16x16x32_f16(false, a, false, b0,
                                                      (short)0, acc0, false, false);
        acc1 = __builtin_amdgcn_wmma_f32_16x16x32_f16(false, a, false, b1,
                                                      (short)0, acc1, false, false);
    }

    // ---- epilogue: two 16x16 tiles ----
#pragma unroll
    for (int t = 0; t < 2; ++t) {
        int col = (t == 0 ? nbase0 : nbase1) + rr;
        if (col >= Cout) continue;
        const v8f& acc = (t == 0) ? acc0 : acc1;
        float bia = bias[col];
        float sc = 1.f, sh = 0.f;
        if (fuse) {
            sc = bng[col] * rsqrtf(bnv[col] + 1e-5f);
            sh = bnb[col] - bnm[col] * sc;
        }
#pragma unroll
        for (int r = 0; r < 8; ++r) {
            int row = mbase + r + 8 * hl;   // C/D layout: lanes>=16 hold M=r+8
            if (row < M) {
                float y = acc[r] + bia;
                if (fuse) y = fmaxf(y * sc + sh, 0.f);
                Y[(size_t)row * Cout + col] = y;
            }
        }
    }
}

// ---------------------------------------------------------------------------
// [B,C,N] -> [B,N,C]
// ---------------------------------------------------------------------------
__global__ void transpose_bcn_bnc_k(const float* __restrict__ in,
                                    float* __restrict__ out,
                                    int B_, int C, int Nn)
{
    int i = blockIdx.x * blockDim.x + threadIdx.x;
    if (i >= B_ * C * Nn) return;
    int n = i % Nn; int t = i / Nn; int c = t % C; int b = t / C;
    out[((size_t)b * Nn + n) * C + c] = in[i];
}

// [B,N,C] -> [B,C,N] (pack to output)
__global__ void transpose_bnc_bcn_k(const float* __restrict__ in,
                                    float* __restrict__ out,
                                    int B_, int Nn, int C)
{
    int i = blockIdx.x * blockDim.x + threadIdx.x;
    if (i >= B_ * Nn * C) return;
    int c = i % C; int t = i / C; int n = t % Nn; int b = t / Nn;
    out[((size_t)b * C + c) * Nn + n] = in[i];
}

// strided channel copy: dst[row*Cd + doff + c] = src[row*Cs + soff + c]
__global__ void copy_ch_k(const float* __restrict__ src, float* __restrict__ dst,
                          int M, int Cs, int soff, int nc, int Cd, int doff)
{
    int i = blockIdx.x * blockDim.x + threadIdx.x;
    if (i >= M * nc) return;
    int row = i / nc, c = i % nc;
    dst[(size_t)row * Cd + doff + c] = src[(size_t)row * Cs + soff + c];
}

// ---------------------------------------------------------------------------
// Farthest point sampling: 1 block (256 threads) per batch.
// ---------------------------------------------------------------------------
__global__ void fps_k(const float* __restrict__ xyz, int* __restrict__ idx,
                      int Nn, int npoint)
{
    extern __shared__ float smem[];
    float* dist = smem;                 // Nn floats
    float* rd   = smem + Nn;            // 256 floats
    int*   ri   = (int*)(rd + 256);     // 256 ints

    const int b = blockIdx.x;
    const int t = threadIdx.x;
    const float* pts = xyz + (size_t)b * Nn * 3;

    for (int i = t; i < Nn; i += 256) dist[i] = 1e10f;
    __syncthreads();

    int far = 0;
    for (int it = 0; it < npoint; ++it) {
        if (t == 0) idx[(size_t)b * npoint + it] = far;
        float fx = pts[far * 3], fy = pts[far * 3 + 1], fz = pts[far * 3 + 2];
        float best = -1.f; int bi = 0;
        for (int i = t; i < Nn; i += 256) {
            float dx = pts[i * 3] - fx, dy = pts[i * 3 + 1] - fy, dz = pts[i * 3 + 2] - fz;
            float d = dx * dx + dy * dy + dz * dz;
            float nd = fminf(dist[i], d);
            dist[i] = nd;
            if (nd > best) { best = nd; bi = i; }
        }
        rd[t] = best; ri[t] = bi;
        __syncthreads();
        for (int s = 128; s > 0; s >>= 1) {
            if (t < s) {
                if (rd[t + s] > rd[t] || (rd[t + s] == rd[t] && ri[t + s] < ri[t])) {
                    rd[t] = rd[t + s]; ri[t] = ri[t + s];
                }
            }
            __syncthreads();
        }
        far = ri[0];
        __syncthreads();
    }
}

// gather xyz rows: out[b,s,:] = xyz[b, idx[b,s], :]
__global__ void gather3_k(const float* __restrict__ xyz, const int* __restrict__ idx,
                          float* __restrict__ out, int Nn, int S)
{
    int b = blockIdx.y;
    int s = blockIdx.x * blockDim.x + threadIdx.x;
    if (s >= S) return;
    int i = idx[(size_t)b * S + s];
    const float* p = xyz + ((size_t)b * Nn + i) * 3;
    float* o = out + ((size_t)b * S + s) * 3;
    o[0] = p[0]; o[1] = p[1]; o[2] = p[2];
}

// ---------------------------------------------------------------------------
// Ball query + group: g[b,s,k, 0..C-1]=points[idx], g[...,C..C+2]=xyz-center
// Fill policy matches reference (first in-radius index repeated).
// ---------------------------------------------------------------------------
__global__ void ball_group_k(const float* __restrict__ xyz,
                             const float* __restrict__ points,
                             const float* __restrict__ newxyz,
                             float* __restrict__ g,
                             int Nn, int S, int C, int K, float r2)
{
    int b = blockIdx.y;
    int s = blockIdx.x * blockDim.x + threadIdx.x;
    if (s >= S) return;
    const float* nz = newxyz + ((size_t)b * S + s) * 3;
    float cx = nz[0], cy = nz[1], cz = nz[2];
    const float* xb = xyz + (size_t)b * Nn * 3;
    const float* pb = points + (size_t)b * Nn * C;
    float* gb = g + ((size_t)b * S + s) * K * (C + 3);

    int cnt = 0, first = -1;
    for (int i = 0; i < Nn && cnt < K; ++i) {
        float dx = xb[i * 3] - cx, dy = xb[i * 3 + 1] - cy, dz = xb[i * 3 + 2] - cz;
        float d = dx * dx + dy * dy + dz * dz;
        if (d <= r2) {
            if (first < 0) first = i;
            float* row = gb + (size_t)cnt * (C + 3);
            for (int c = 0; c < C; ++c) row[c] = pb[(size_t)i * C + c];
            row[C] = dx; row[C + 1] = dy; row[C + 2] = dz;
            cnt++;
        }
    }
    if (first < 0) first = 0;
    for (int k = cnt; k < K; ++k) {
        float* row = gb + (size_t)k * (C + 3);
        if (cnt > 0) {
            for (int c = 0; c < C + 3; ++c) row[c] = gb[c];
        } else {
            int i = first;
            for (int c = 0; c < C; ++c) row[c] = pb[(size_t)i * C + c];
            row[C] = xb[i * 3] - cx; row[C + 1] = xb[i * 3 + 1] - cy; row[C + 2] = xb[i * 3 + 2] - cz;
        }
    }
}

// max over K: x [BS, K, C] -> y[row*Cd + off + c]
__global__ void maxpool_k(const float* __restrict__ x, float* __restrict__ y,
                          int BS, int K, int C, int Cd, int off)
{
    int i = blockIdx.x * blockDim.x + threadIdx.x;
    if (i >= BS * C) return;
    int row = i / C, c = i % C;
    const float* p = x + ((size_t)row * K) * C + c;
    float m = p[0];
    for (int k = 1; k < K; ++k) m = fmaxf(m, p[(size_t)k * C]);
    y[(size_t)row * Cd + off + c] = m;
}

// 3-NN inverse-distance interpolation into dst[...,doff..doff+C2)
__global__ void interp3_k(const float* __restrict__ xyz1, const float* __restrict__ xyz2,
                          const float* __restrict__ p2, float* __restrict__ dst,
                          int N1, int N2, int C2, int Cd, int doff)
{
    int b = blockIdx.y;
    int n = blockIdx.x * blockDim.x + threadIdx.x;
    if (n >= N1) return;
    const float* q = xyz1 + ((size_t)b * N1 + n) * 3;
    float qx = q[0], qy = q[1], qz = q[2];
    float d0 = 1e30f, d1 = 1e30f, d2 = 1e30f;
    int i0 = 0, i1 = 0, i2 = 0;
    for (int j = 0; j < N2; ++j) {
        const float* p = xyz2 + ((size_t)b * N2 + j) * 3;
        float dx = p[0] - qx, dy = p[1] - qy, dz = p[2] - qz;
        float d = dx * dx + dy * dy + dz * dz;
        if (d < d0)      { d2 = d1; i2 = i1; d1 = d0; i1 = i0; d0 = d; i0 = j; }
        else if (d < d1) { d2 = d1; i2 = i1; d1 = d;  i1 = j; }
        else if (d < d2) { d2 = d;  i2 = j; }
    }
    float w0 = 1.f / (d0 + 1e-8f), w1 = 1.f / (d1 + 1e-8f), w2 = 1.f / (d2 + 1e-8f);
    float ws = w0 + w1 + w2;
    w0 /= ws; w1 /= ws; w2 /= ws;
    const float* pbase = p2 + (size_t)b * N2 * C2;
    float* o = dst + ((size_t)b * N1 + n) * Cd + doff;
    for (int c = 0; c < C2; ++c)
        o[c] = w0 * pbase[(size_t)i0 * C2 + c] + w1 * pbase[(size_t)i1 * C2 + c]
             + w2 * pbase[(size_t)i2 * C2 + c];
}

// standalone BN + ReLU
__global__ void bn_relu_k(const float* __restrict__ x, float* __restrict__ y,
                          const float* __restrict__ g, const float* __restrict__ bb,
                          const float* __restrict__ m, const float* __restrict__ v,
                          int M, int C)
{
    int i = blockIdx.x * blockDim.x + threadIdx.x;
    if (i >= M * C) return;
    int c = i % C;
    float sc = g[c] * rsqrtf(v[c] + 1e-5f);
    y[i] = fmaxf((x[i] - m[c]) * sc + bb[c], 0.f);
}

// row-wise log_softmax over last dim C (C small: <=50)
__global__ void log_softmax_k(const float* __restrict__ x, float* __restrict__ y,
                              int M, int C)
{
    int r = blockIdx.x * blockDim.x + threadIdx.x;
    if (r >= M) return;
    const float* p = x + (size_t)r * C;
    float mx = p[0];
    for (int c = 1; c < C; ++c) mx = fmaxf(mx, p[c]);
    float s = 0.f;
    for (int c = 0; c < C; ++c) s += __expf(p[c] - mx);
    float l = __logf(s);
    float* o = y + (size_t)r * C;
    for (int c = 0; c < C; ++c) o[c] = p[c] - mx - l;
}

// ---------------------------------------------------------------------------
// Host-side orchestration
// ---------------------------------------------------------------------------
struct BNP { const float *b, *g, *m, *v; };
struct LYR { const float *W, *bi; BNP bn; };

extern "C" void kernel_launch(void* const* d_in, const int* in_sizes, int n_in,
                              void* d_out, int out_size, void* d_ws, size_t ws_size,
                              hipStream_t stream)
{
    (void)in_sizes; (void)n_in; (void)out_size; (void)ws_size;
    const float* xyz_in = (const float*)d_in[0];   // [B, 9, N]

    // ---- parse params (jax pytree leaf order: sorted dict keys) ----
    int cur = 1;
    auto nf = [&]() -> const float* { return (const float*)d_in[cur++]; };
    auto rdBN  = [&](BNP& p) { p.b = nf(); p.g = nf(); p.m = nf(); p.v = nf(); };
    auto rdLyr = [&](LYR* a, int n) {
        for (int i = 0; i < n; ++i) { a[i].W = nf(); a[i].bi = nf(); rdBN(a[i].bn); }
    };

    BNP BNfc, BN1, BN2;
    rdBN(BNfc); rdBN(BN1); rdBN(BN2);
    const float *c1W = nf(), *c1b = nf();
    const float *c2W = nf(), *c2b = nf();
    const float *c3W = nf(), *c3b = nf();
    const float *fcW = nf(), *fcb = nf();
    const float *f2W = nf(), *f2b = nf();
    LYR FP1[3], FP2[2], FP3[2], FP4[2];
    rdLyr(FP1, 3); rdLyr(FP2, 2); rdLyr(FP3, 2); rdLyr(FP4, 2);
    LYR SA1a[3], SA1b[3], SA2a[3], SA2b[3], SA3a[3], SA3b[3], SA4a[3], SA4b[3], SA5[3];
    rdLyr(SA1a, 3); rdLyr(SA1b, 3);
    rdLyr(SA2a, 3); rdLyr(SA2b, 3);
    rdLyr(SA3a, 3); rdLyr(SA3b, 3);
    rdLyr(SA4a, 3); rdLyr(SA4b, 3);
    rdLyr(SA5, 3);

    // ---- workspace bump allocator ----
    float* wsf = (float*)d_ws;
    size_t woff = 0;
    auto alloc = [&](size_t n) -> float* {
        float* p = wsf + woff;
        woff += (n + 63) & ~(size_t)63;
        return p;
    };
    float* x      = alloc((size_t)BB * NN * 9);     // l0_p
    float* l0xyz  = alloc((size_t)BB * NN * 3);
    int*   fidx   = (int*)alloc(BB * 1024);
    float* l1xyz  = alloc(BB * 1024 * 3);
    float* l1p    = alloc((size_t)BB * 1024 * 96);
    float* l2xyz  = alloc(BB * 256 * 3);
    float* l2p    = alloc((size_t)BB * 256 * 256);
    float* l3xyz  = alloc(BB * 64 * 3);
    float* l3p    = alloc((size_t)BB * 64 * 512);
    float* l4xyz  = alloc(BB * 16 * 3);
    float* l4p    = alloc((size_t)BB * 16 * 1024);
    float* l5p    = alloc(BB * 1024);
    float* l3p2   = alloc((size_t)BB * 64 * 256);
    float* l2p2   = alloc((size_t)BB * 256 * 256);
    float* l1p2   = alloc((size_t)BB * 1024 * 128);
    float* l0out  = alloc((size_t)BB * NN * 128);
    float* x1buf  = alloc((size_t)BB * NN * 50);
    float* x2buf  = alloc((size_t)BB * NN * 50);
    float* x3buf  = alloc((size_t)BB * NN * 50);
    float* pingA  = alloc((size_t)8 * 1024 * 32 * 32);   // 8.39M floats
    float* pingB  = alloc((size_t)8 * 1024 * 32 * 64);   // 16.78M floats

    auto gemm = [&](const float* X, const float* W, const float* bias, const BNP* bn,
                    float* Y, int M, int Ci, int Co) {
        dim3 g((M + 63) / 64, (Co + 31) / 32);
        if ((Ci & 3) == 0)
            gemm_bn_relu_k<true><<<g, 128, 0, stream>>>(X, W, bias,
                bn ? bn->g : bias, bn ? bn->b : bias, bn ? bn->m : bias, bn ? bn->v : bias,
                Y, M, Ci, Co, bn ? 1 : 0);
        else
            gemm_bn_relu_k<false><<<g, 128, 0, stream>>>(X, W, bias,
                bn ? bn->g : bias, bn ? bn->b : bias, bn ? bn->m : bias, bn ? bn->v : bias,
                Y, M, Ci, Co, bn ? 1 : 0);
    };
    auto run_mlp = [&](float* in, int M, int Ci, const LYR* L, const int* ch, int nl) -> float* {
        const float* curp = in;
        float* out = nullptr;
        int ci = Ci;
        for (int i = 0; i < nl; ++i) {
            out = (curp == pingB) ? pingA : pingB;
            gemm(curp, L[i].W, L[i].bi, &L[i].bn, out, M, ci, ch[i]);
            curp = out; ci = ch[i];
        }
        return out;
    };
    auto sa_branch = [&](const float* xyzl, const float* pts, int Np, int C,
                         const float* nxyz, int S, float radius, int K,
                         const LYR* L, const int* ch, float* outp, int Cd, int doff) {
        dim3 gb((S + 127) / 128, BB);
        ball_group_k<<<gb, 128, 0, stream>>>(xyzl, pts, nxyz, pingA, Np, S, C, K,
                                             radius * radius);
        int M = BB * S * K;
        float* last = run_mlp(pingA, M, C + 3, L, ch, 3);
        int BS = BB * S, co = ch[2], tot = BS * co;
        maxpool_k<<<(tot + 255) / 256, 256, 0, stream>>>(last, outp, BS, K, co, Cd, doff);
    };
    auto sa_msg = [&](const float* xyzl, const float* pts, int Np, int C, int npoint,
                      float r1, int K1, const LYR* L1, const int* ch1,
                      float r2, int K2, const LYR* L2, const int* ch2,
                      float* nxyz, float* outp) {
        fps_k<<<BB, 256, (size_t)(Np + 512) * 4, stream>>>(xyzl, fidx, Np, npoint);
        dim3 gg((npoint + 127) / 128, BB);
        gather3_k<<<gg, 128, 0, stream>>>(xyzl, fidx, nxyz, Np, npoint);
        int Cd = ch1[2] + ch2[2];
        sa_branch(xyzl, pts, Np, C, nxyz, npoint, r1, K1, L1, ch1, outp, Cd, 0);
        sa_branch(xyzl, pts, Np, C, nxyz, npoint, r2, K2, L2, ch2, outp, Cd, ch1[2]);
    };
    auto fp = [&](const float* xyz1, int N1, const float* xyz2, int N2,
                  const float* p1, int C1, const float* p2, int C2,
                  const LYR* L, const int* ch, int nl, float* dest) {
        int M = BB * N1, Cd = C1 + C2;
        if (p1) {
            int tot = M * C1;
            copy_ch_k<<<(tot + 255) / 256, 256, 0, stream>>>(p1, pingA, M, C1, 0, C1, Cd, 0);
        }
        dim3 gi((N1 + 127) / 128, BB);
        interp3_k<<<gi, 128, 0, stream>>>(xyz1, xyz2, p2, pingA, N1, N2, C2, Cd, C1);
        float* last = run_mlp(pingA, M, Cd, L, ch, nl);
        int co = ch[nl - 1], tot = M * co;
        copy_ch_k<<<(tot + 255) / 256, 256, 0, stream>>>(last, dest, M, co, 0, co, co, 0);
    };

    // ---- forward pass ----
    {   // transpose input [B,9,N] -> x [B,N,9]; slice l0_xyz
        int tot = BB * 9 * NN;
        transpose_bcn_bnc_k<<<(tot + 255) / 256, 256, 0, stream>>>(xyz_in, x, BB, 9, NN);
        tot = BB * NN * 3;
        copy_ch_k<<<(tot + 255) / 256, 256, 0, stream>>>(x, l0xyz, BB * NN, 9, 0, 3, 3, 0);
    }
    static const int sa1c1[3] = {16, 16, 32},  sa1c2[3] = {32, 32, 64};
    static const int sa2c1[3] = {64, 64, 128}, sa2c2[3] = {64, 96, 128};
    static const int sa3c[3]  = {128, 196, 256};
    static const int sa4c1[3] = {256, 256, 512}, sa4c2[3] = {256, 384, 512};
    static const int sa5c[3]  = {256, 512, 1024};
    static const int fp4c[2]  = {256, 256}, fp3c[2] = {256, 256}, fp2c[2] = {256, 128};
    static const int fp1c[3]  = {128, 128, 128};

    sa_msg(l0xyz, x,    NN,   9,   1024, 0.05f, 16, SA1a, sa1c1, 0.1f, 32, SA1b, sa1c2, l1xyz, l1p);
    sa_msg(l1xyz, l1p,  1024, 96,  256,  0.1f,  16, SA2a, sa2c1, 0.2f, 32, SA2b, sa2c2, l2xyz, l2p);
    sa_msg(l2xyz, l2p,  256,  256, 64,   0.2f,  16, SA3a, sa3c,  0.4f, 32, SA3b, sa3c,  l3xyz, l3p);
    sa_msg(l3xyz, l3p,  64,   512, 16,   0.4f,  16, SA4a, sa4c1, 0.8f, 32, SA4b, sa4c2, l4xyz, l4p);

    {   // sa5: concat [xyz, points] over 16 points, MLP, max -> l5p [B,1024]
        int M5 = BB * 16;
        int tot = M5 * 3;
        copy_ch_k<<<(tot + 255) / 256, 256, 0, stream>>>(l4xyz, pingA, M5, 3, 0, 3, 1027, 0);
        tot = M5 * 1024;
        copy_ch_k<<<(tot + 255) / 256, 256, 0, stream>>>(l4p, pingA, M5, 1024, 0, 1024, 1027, 3);
        float* last = run_mlp(pingA, M5, 1027, SA5, sa5c, 3);
        tot = BB * 1024;
        maxpool_k<<<(tot + 255) / 256, 256, 0, stream>>>(last, l5p, BB, 16, 1024, 1024, 0);
    }

    fp(l3xyz, 64,   l4xyz, 16,  l3p,   512, l4p,  1024, FP4, fp4c, 2, l3p2);
    fp(l2xyz, 256,  l3xyz, 64,  l2p,   256, l3p2, 256,  FP3, fp3c, 2, l2p2);
    fp(l1xyz, 1024, l2xyz, 256, l1p,   96,  l2p2, 256,  FP2, fp2c, 2, l1p2);
    fp(l0xyz, NN,   l1xyz, 1024, nullptr, 0, l1p2, 128, FP1, fp1c, 3, l0out);

    float* out = (float*)d_out;
    const size_t o0 = 0;
    const size_t o1 = (size_t)BB * 128 * NN;
    const size_t o2 = o1 + (size_t)BB * 50 * NN;
    const size_t o3 = o2 + (size_t)BB * 50 * NN;
    const size_t o4 = o3 + (size_t)BB * 50 * NN;

    {   // classification head: relu(bn(fc(l5))) -> fc2 -> log_softmax -> out[o4]
        gemm(l5p, fcW, fcb, &BNfc, pingA, BB, 1024, 256);
        gemm(pingA, f2W, f2b, nullptr, pingB, BB, 256, 16);
        log_softmax_k<<<1, 32, 0, stream>>>(pingB, out + o4, BB, 16);
    }
    {   // segmentation heads
        int M = BB * NN;
        gemm(l0out, c1W, c1b, nullptr, pingA, M, 128, 50);
        log_softmax_k<<<(M + 127) / 128, 128, 0, stream>>>(pingA, x1buf, M, 50);
        int tot = M * 50;
        bn_relu_k<<<(tot + 255) / 256, 256, 0, stream>>>(x1buf, pingB, BN1.g, BN1.b, BN1.m, BN1.v, M, 50);
        gemm(pingB, c2W, c2b, nullptr, pingA, M, 50, 50);
        log_softmax_k<<<(M + 127) / 128, 128, 0, stream>>>(pingA, x2buf, M, 50);
        bn_relu_k<<<(tot + 255) / 256, 256, 0, stream>>>(x2buf, pingB, BN2.g, BN2.b, BN2.m, BN2.v, M, 50);
        gemm(pingB, c3W, c3b, nullptr, pingA, M, 50, 50);
        log_softmax_k<<<(M + 127) / 128, 128, 0, stream>>>(pingA, x3buf, M, 50);
    }
    {   // pack transposed outputs
        int tot = BB * NN * 128;
        transpose_bnc_bcn_k<<<(tot + 255) / 256, 256, 0, stream>>>(l0out, out + o0, BB, NN, 128);
        tot = BB * NN * 50;
        transpose_bnc_bcn_k<<<(tot + 255) / 256, 256, 0, stream>>>(x1buf, out + o1, BB, NN, 50);
        transpose_bnc_bcn_k<<<(tot + 255) / 256, 256, 0, stream>>>(x2buf, out + o2, BB, NN, 50);
        transpose_bnc_bcn_k<<<(tot + 255) / 256, 256, 0, stream>>>(x3buf, out + o3, BB, NN, 50);
    }
}
